// Attention_81432579932345
// MI455X (gfx1250) — compile-verified, matrix operand order checked by reading, not run on hardware
//
#include <hip/hip_runtime.h>
#include <hip/hip_bf16.h>

// Problem constants (fixed by the reference)
#define S_LEN 4096
#define HID 1024
#define NH 16
#define NKV 4
#define HD 64
#define NREP (NH / NKV)
#define WINDOW 512
#define KVSTRIDE (NKV * HD)   // 256

typedef __attribute__((ext_vector_type(16))) _Float16 v16h;
typedef __attribute__((ext_vector_type(8)))  float    v8f;

// ---------------------------------------------------------------------------
// WMMA wrapper: D = A(16x32 f16) * B(32x16 f16) + C(16x16 f32)
// ---------------------------------------------------------------------------
__device__ __forceinline__ v8f wmma16x16x32(v16h a, v16h b, v8f c) {
    return __builtin_amdgcn_wmma_f32_16x16x32_f16(
        /*neg_a=*/false, a, /*neg_b=*/false, b,
        /*c_mod=*/(short)0, c, /*reuse_a=*/false, /*reuse_b=*/false);
}

// Fragment loader, K contiguous, f16 source (global or LDS).
// ISA 16-bit A/B layout: elements 0..7 hold K = base+off+0..7, elements 8..15
// hold K = base+off+16..23, off = (lane>=16)?8:0 (caller folds off into p).
// Each 8-half chunk is 16 contiguous bytes -> one b128 load.
__device__ __forceinline__ v16h load_frag16(const _Float16* p) {
    v16h r;
#pragma unroll
    for (int e = 0; e < 8; ++e) {
        r[e]     = p[e];
        r[8 + e] = p[16 + e];
    }
    return r;
}

// V B-fragment out of the LDS tile Vt[32 keys][64 dims]: lane holds one output
// dim (dcol), elements stride over local key index (K of the matmul).
__device__ __forceinline__ v16h load_vfrag_lds(const _Float16* vt, int dcol, int off) {
    v16h r;
#pragma unroll
    for (int e = 0; e < 8; ++e) {
        r[e]     = vt[(off + e) * HD + dcol];
        r[8 + e] = vt[(16 + off + e) * HD + dcol];
    }
    return r;
}

__device__ __forceinline__ float red_max16(float x) {
#pragma unroll
    for (int m = 1; m < 16; m <<= 1) x = fmaxf(x, __shfl_xor(x, m, 32));
    return x;
}
__device__ __forceinline__ float red_sum16(float x) {
#pragma unroll
    for (int m = 1; m < 16; m <<= 1) x += __shfl_xor(x, m, 32);
    return x;
}

// ---------------------------------------------------------------------------
// Elementwise f32 -> f16 conversion (one-time pre-pass; removes cvts and
// halves bytes in all WMMA inner loops).
// ---------------------------------------------------------------------------
__global__ void __launch_bounds__(256)
cvt_f32_to_f16(const float* __restrict__ in, _Float16* __restrict__ out, int n) {
    int i = blockIdx.x * 256 + threadIdx.x;
    if (i < n) out[i] = (_Float16)in[i];
}

// ---------------------------------------------------------------------------
// GEMM: C[M,N] = A16[M,K] * W16[N,K]^T, all-f16 operands, f32 accumulate.
// One wave computes a 32x64 tile: 2 A-frags x 4 B-frags = 8 WMMAs / K-step,
// each B fragment reused twice. grid = (M/32, N/64), block = 32 (wave32).
// ---------------------------------------------------------------------------
__global__ void __launch_bounds__(32)
gemm_f16_32x64(const _Float16* __restrict__ A, const _Float16* __restrict__ W,
               float* __restrict__ C, int M, int N, int K) {
    const int lane = threadIdx.x;
    const int nlo  = lane & 15;
    const int hi   = (lane >> 4) & 1;
    const int off  = hi * 8;
    const int m0   = blockIdx.x * 32;
    const int n0   = blockIdx.y * 64;

    const v8f z = {};
    v8f acc[2][4];
#pragma unroll
    for (int i = 0; i < 2; ++i)
#pragma unroll
        for (int t = 0; t < 4; ++t) acc[i][t] = z;

    const _Float16* a0 = A + (size_t)(m0 + nlo) * K;
    const _Float16* a1 = A + (size_t)(m0 + 16 + nlo) * K;
    const _Float16* w0 = W + (size_t)(n0 +  0 + nlo) * K;
    const _Float16* w1 = W + (size_t)(n0 + 16 + nlo) * K;
    const _Float16* w2 = W + (size_t)(n0 + 32 + nlo) * K;
    const _Float16* w3 = W + (size_t)(n0 + 48 + nlo) * K;

    for (int kb = 0; kb < K; kb += 32) {
        // speculative prefetch of the next K-block (global_prefetch_b8)
        __builtin_prefetch(a0 + kb + 32, 0, 1);
        __builtin_prefetch(w0 + kb + 32, 0, 1);

        v16h fa0 = load_frag16(a0 + kb + off);
        v16h fa1 = load_frag16(a1 + kb + off);

        v16h fb = load_frag16(w0 + kb + off);
        acc[0][0] = wmma16x16x32(fa0, fb, acc[0][0]);
        acc[1][0] = wmma16x16x32(fa1, fb, acc[1][0]);
        fb = load_frag16(w1 + kb + off);
        acc[0][1] = wmma16x16x32(fa0, fb, acc[0][1]);
        acc[1][1] = wmma16x16x32(fa1, fb, acc[1][1]);
        fb = load_frag16(w2 + kb + off);
        acc[0][2] = wmma16x16x32(fa0, fb, acc[0][2]);
        acc[1][2] = wmma16x16x32(fa1, fb, acc[1][2]);
        fb = load_frag16(w3 + kb + off);
        acc[0][3] = wmma16x16x32(fa0, fb, acc[0][3]);
        acc[1][3] = wmma16x16x32(fa1, fb, acc[1][3]);
    }

#pragma unroll
    for (int i = 0; i < 2; ++i) {
#pragma unroll
        for (int r = 0; r < 8; ++r) {
            int row = m0 + 16 * i + r + hi * 8;
            float* cp = C + (size_t)row * N + n0 + nlo;
            cp[0]  = acc[i][0][r];
            cp[16] = acc[i][1][r];
            cp[32] = acc[i][2][r];
            cp[48] = acc[i][3][r];
        }
    }
}

// ---------------------------------------------------------------------------
// Per-head RMSNorm + RoPE: read f32 projections, emit f16 heads for the
// attention kernel. Lane d handles dims d and d+32 (rotate_half pairing).
// grid = (S, NH+NKV), block = 32.
// ---------------------------------------------------------------------------
__global__ void __launch_bounds__(32)
norm_rope_kernel(const float* __restrict__ qbuf, const float* __restrict__ kbuf,
                 _Float16* __restrict__ q16, _Float16* __restrict__ k16,
                 const float* __restrict__ cosb, const float* __restrict__ sinb,
                 const float* __restrict__ qw, const float* __restrict__ kw) {
    const int s  = blockIdx.x;
    const int hh = blockIdx.y;
    const int d  = threadIdx.x;           // 0..31

    const float* src;
    _Float16* dst;
    const float* w;
    if (hh < NH) {
        src = qbuf + (size_t)s * (NH * HD) + hh * HD;
        dst = q16  + (size_t)s * (NH * HD) + hh * HD;
        w = qw;
    } else {
        src = kbuf + (size_t)s * KVSTRIDE + (hh - NH) * HD;
        dst = k16  + (size_t)s * KVSTRIDE + (hh - NH) * HD;
        w = kw;
    }

    float v0 = src[d], v1 = src[d + 32];
    float ss = v0 * v0 + v1 * v1;
#pragma unroll
    for (int m = 1; m < 32; m <<= 1) ss += __shfl_xor(ss, m, 32);
    float rn = rsqrtf(ss * (1.0f / (float)HD) + 1e-5f);
    float n0 = v0 * rn * w[d];
    float n1 = v1 * rn * w[d + 32];

    float c0 = cosb[(size_t)s * HD + d],      s0 = sinb[(size_t)s * HD + d];
    float c1 = cosb[(size_t)s * HD + d + 32], s1 = sinb[(size_t)s * HD + d + 32];

    dst[d]      = (_Float16)(n0 * c0 - n1 * s0);   // rotate_half: -v[d+32], d<32
    dst[d + 32] = (_Float16)(n1 * c1 + n0 * s1);   //              +v[d],    d>=32
}

// ---------------------------------------------------------------------------
// Flash attention, sliding window + sink. One wave per (16-query tile, head).
// K/V tiles (32 keys x 64 dims, f16) are staged global->LDS with the CDNA5
// async-copy path (GLOBAL_LOAD_ASYNC_TO_LDS_B128 + s_wait_asynccnt), then
// 4 score WMMAs (QK^T, K=64) + 4 PV WMMAs (D=64) per 32-key block.
// P (prob tile) bounced through LDS to convert C-layout -> A-layout.
// Output written directly as f16 for the wo projection.
// grid = (S/16, NH), block = 32.
// ---------------------------------------------------------------------------
__global__ void __launch_bounds__(32)
attn_kernel(const _Float16* __restrict__ q16, const _Float16* __restrict__ k16,
            const _Float16* __restrict__ v16, const float* __restrict__ sinks,
            _Float16* __restrict__ o16) {
    __shared__ _Float16 Kt[32 * HD];      // [key_local][d]
    __shared__ _Float16 Vt[32 * HD];      // [key_local][d]
    __shared__ _Float16 P[16 * 32];       // [row][key_local]

    const int lane = threadIdx.x;
    const int nlo  = lane & 15;
    const int hi   = (lane >> 4) & 1;
    const int off  = hi * 8;
    const int q0   = blockIdx.x * 16;
    const int h    = blockIdx.y;
    const int kvh  = h / NREP;
    const float scale = 0.125f;           // 1/sqrt(64)

    // Q A-fragments for d-chunks [0,32) and [32,64)
    const _Float16* qp = q16 + (size_t)(q0 + nlo) * HID + h * HD;
    const v16h aq0 = load_frag16(qp + off);
    const v16h aq1 = load_frag16(qp + 32 + off);

    float mrow[8], lrow[8];
    v8f o0 = {}, o1 = {}, o2 = {}, o3 = {};
#pragma unroll
    for (int r = 0; r < 8; ++r) { mrow[r] = -1e30f; lrow[r] = 0.0f; }

    const unsigned kt_base = (unsigned)(uintptr_t)(&Kt[0]) + (unsigned)lane * (HD * 2);
    const unsigned vt_base = (unsigned)(uintptr_t)(&Vt[0]) + (unsigned)lane * (HD * 2);

    const int qend = q0 + 15;
    int kstart = q0 - (WINDOW - 1);
    if (kstart < 0) kstart = 0;
    kstart &= ~31;

    for (int j0 = kstart; j0 <= qend; j0 += 32) {
        // ---- async DMA: stage K/V 32x64 f16 tiles into LDS (ASYNCcnt) ----
        {
            int key = j0 + lane;
            if (key > S_LEN - 1) key = S_LEN - 1;       // tail rows masked below
            const _Float16* kg = k16 + (size_t)key * KVSTRIDE + kvh * HD;
            const _Float16* vg = v16 + (size_t)key * KVSTRIDE + kvh * HD;
#pragma unroll
            for (int c = 0; c < 8; ++c) {               // 8 x 16B per 128B row
                unsigned long long ka = (unsigned long long)(uintptr_t)kg + c * 16u;
                unsigned long long va = (unsigned long long)(uintptr_t)vg + c * 16u;
                asm volatile("global_load_async_to_lds_b128 %0, %1, off"
                             :: "v"(kt_base + c * 16u), "v"(ka) : "memory");
                asm volatile("global_load_async_to_lds_b128 %0, %1, off"
                             :: "v"(vt_base + c * 16u), "v"(va) : "memory");
            }
            asm volatile("s_wait_asynccnt 0x0" ::: "memory");
        }

        // ---- scores: two 16-key tiles out of Kt, K=64 in two 32-chunks ----
        const _Float16* kl0 = Kt + nlo * HD;
        const _Float16* kl1 = Kt + (16 + nlo) * HD;
        v8f c0 = {}, c1 = {};
        c0 = wmma16x16x32(aq0, load_frag16(kl0 + off),      c0);
        c0 = wmma16x16x32(aq1, load_frag16(kl0 + 32 + off), c0);
        c1 = wmma16x16x32(aq0, load_frag16(kl1 + off),      c1);
        c1 = wmma16x16x32(aq1, load_frag16(kl1 + 32 + off), c1);

        // ---- mask + online softmax (per row; 16-lane shuffle reductions) ----
#pragma unroll
        for (int r = 0; r < 8; ++r) {
            const int i  = q0 + r + hi * 8;
            const int ja = j0 + nlo;
            const int jb = j0 + 16 + nlo;
            float s0 = ((ja <= i) && (ja > i - WINDOW)) ? c0[r] * scale : -1e30f;
            float s1 = ((jb <= i) && (jb > i - WINDOW)) ? c1[r] * scale : -1e30f;

            float rm = red_max16(fmaxf(s0, s1));
            float nm = fmaxf(mrow[r], rm);
            float alpha = __expf(mrow[r] - nm);
            float p0 = (s0 <= -1e29f) ? 0.0f : __expf(s0 - nm);
            float p1 = (s1 <= -1e29f) ? 0.0f : __expf(s1 - nm);
            float rs = red_sum16(p0 + p1);
            lrow[r] = lrow[r] * alpha + rs;
            mrow[r] = nm;
            o0[r] *= alpha; o1[r] *= alpha; o2[r] *= alpha; o3[r] *= alpha;

            const int row = r + hi * 8;
            P[row * 32 + nlo]      = (_Float16)p0;
            P[row * 32 + 16 + nlo] = (_Float16)p1;
        }

        // ---- P (16x32) as A fragment, V tile from LDS, 4 PV WMMAs ----
        v16h pa = load_frag16(P + nlo * 32 + off);
        o0 = wmma16x16x32(pa, load_vfrag_lds(Vt,  0 + nlo, off), o0);
        o1 = wmma16x16x32(pa, load_vfrag_lds(Vt, 16 + nlo, off), o1);
        o2 = wmma16x16x32(pa, load_vfrag_lds(Vt, 32 + nlo, off), o2);
        o3 = wmma16x16x32(pa, load_vfrag_lds(Vt, 48 + nlo, off), o3);
    }

    // ---- fold sink into denominator, normalize, write f16 [S, H*D] ----
    const float snk = sinks[h];
#pragma unroll
    for (int r = 0; r < 8; ++r) {
        float nm    = fmaxf(mrow[r], snk);
        float corr  = __expf(mrow[r] - nm);
        float denom = lrow[r] * corr + __expf(snk - nm);
        float sc    = corr / denom;
        const int row = q0 + r + hi * 8;
        _Float16* op = o16 + (size_t)row * HID + h * HD + nlo;
        op[0]  = (_Float16)(o0[r] * sc);
        op[16] = (_Float16)(o1[r] * sc);
        op[32] = (_Float16)(o2[r] * sc);
        op[48] = (_Float16)(o3[r] * sc);
    }
}

// ---------------------------------------------------------------------------
// Launch: f16 pre-pass -> qkv GEMMs -> norm+rope(f16) -> flash attention ->
// output GEMM. All matmul inner loops are pure-f16 WMMA paths.
// ---------------------------------------------------------------------------
extern "C" void kernel_launch(void* const* d_in, const int* in_sizes, int n_in,
                              void* d_out, int out_size, void* d_ws, size_t ws_size,
                              hipStream_t stream) {
    (void)in_sizes; (void)n_in; (void)out_size; (void)ws_size;
    const float* x     = (const float*)d_in[0];
    const float* cosb  = (const float*)d_in[1];
    const float* sinb  = (const float*)d_in[2];
    const float* wq    = (const float*)d_in[3];
    const float* wk    = (const float*)d_in[4];
    const float* wv    = (const float*)d_in[5];
    const float* wo    = (const float*)d_in[6];
    const float* qnw   = (const float*)d_in[7];
    const float* knw   = (const float*)d_in[8];
    const float* sinks = (const float*)d_in[9];
    float* out = (float*)d_out;

    // workspace carve-up (bytes)
    char* ws = (char*)d_ws;
    _Float16* x16  = (_Float16*)ws;                 ws += (size_t)S_LEN * HID * 2;      // 8 MB
    _Float16* wq16 = (_Float16*)ws;                 ws += (size_t)HID * HID * 2;        // 2 MB
    _Float16* wk16 = (_Float16*)ws;                 ws += (size_t)KVSTRIDE * HID * 2;   // 0.5 MB
    _Float16* wv16 = (_Float16*)ws;                 ws += (size_t)KVSTRIDE * HID * 2;   // 0.5 MB
    _Float16* wo16 = (_Float16*)ws;                 ws += (size_t)HID * HID * 2;        // 2 MB
    float*    qbuf = (float*)ws;                    ws += (size_t)S_LEN * HID * 4;      // 16 MB
    float*    kbuf = (float*)ws;                    ws += (size_t)S_LEN * KVSTRIDE * 4; // 4 MB
    float*    vbuf = (float*)ws;                    ws += (size_t)S_LEN * KVSTRIDE * 4; // 4 MB
    _Float16* q16  = (_Float16*)ws;                 ws += (size_t)S_LEN * HID * 2;      // 8 MB
    _Float16* k16  = (_Float16*)ws;                 ws += (size_t)S_LEN * KVSTRIDE * 2; // 2 MB
    _Float16* v16  = (_Float16*)ws;                 ws += (size_t)S_LEN * KVSTRIDE * 2; // 2 MB
    _Float16* o16  = (_Float16*)qbuf;               // alias: qbuf dead after norm_rope

    // one-time f32 -> f16 conversions
    const int xN  = S_LEN * HID;
    const int wqN = HID * HID;
    const int wkN = KVSTRIDE * HID;
    cvt_f32_to_f16<<<dim3((xN  + 255) / 256), 256, 0, stream>>>(x,  x16,  xN);
    cvt_f32_to_f16<<<dim3((wqN + 255) / 256), 256, 0, stream>>>(wq, wq16, wqN);
    cvt_f32_to_f16<<<dim3((wkN + 255) / 256), 256, 0, stream>>>(wk, wk16, wkN);
    cvt_f32_to_f16<<<dim3((wkN + 255) / 256), 256, 0, stream>>>(wv, wv16, wkN);
    cvt_f32_to_f16<<<dim3((wqN + 255) / 256), 256, 0, stream>>>(wo, wo16, wqN);

    // QKV projections (f16 WMMA, f32 accumulate)
    gemm_f16_32x64<<<dim3(S_LEN / 32, (NH * HD) / 64), 32, 0, stream>>>(x16, wq16, qbuf, S_LEN, NH * HD,  HID);
    gemm_f16_32x64<<<dim3(S_LEN / 32, KVSTRIDE / 64),  32, 0, stream>>>(x16, wk16, kbuf, S_LEN, KVSTRIDE, HID);
    gemm_f16_32x64<<<dim3(S_LEN / 32, KVSTRIDE / 64),  32, 0, stream>>>(x16, wv16, vbuf, S_LEN, KVSTRIDE, HID);

    // RMSNorm + RoPE -> f16 q/k ; v -> f16 passthrough
    norm_rope_kernel<<<dim3(S_LEN, NH + NKV), 32, 0, stream>>>(qbuf, kbuf, q16, k16, cosb, sinb, qnw, knw);
    cvt_f32_to_f16<<<dim3((S_LEN * KVSTRIDE + 255) / 256), 256, 0, stream>>>(vbuf, v16, S_LEN * KVSTRIDE);

    // sliding-window flash attention with sink
    attn_kernel<<<dim3(S_LEN / 16, NH), 32, 0, stream>>>(q16, k16, v16, sinks, o16);

    // output projection
    gemm_f16_32x64<<<dim3(S_LEN / 32, HID / 64), 32, 0, stream>>>(o16, wo16, out, S_LEN, HID, NH * HD);
}
